// MyTransformer_79774722556038
// MI455X (gfx1250) — compile-verified, matrix operand order checked by reading, not run on hardware
//
#include <hip/hip_runtime.h>
#include <hip/hip_bf16.h>

// ---------------------------------------------------------------------------
// MI455X (gfx1250) transformer forward. All GEMMs + attention through
// v_wmma_f32_16x16x32_bf16 (wave32). fp32 master copies, bf16 operands.
// - QKV projection GEMMs emit bf16 directly (V transposed) so the flash
//   inner loop does no format conversion.
// - Flash attention computes S^T = K@Q^T (row-max is in-lane), softmax
//   denominator via ones-matrix WMMA, P re-layout via one-wave LDS bounce.
// - GEMM: ping-pong LDS double buffering, K-step 64, 8 WMMAs per barrier.
// ---------------------------------------------------------------------------

typedef __bf16 bf16_t;
typedef bf16_t v16bf __attribute__((ext_vector_type(16)));
typedef float  v8f   __attribute__((ext_vector_type(8)));

union Frag {
    v16bf v;
    uint4 q[2];
    unsigned short u[16];
    bf16_t h[16];
};
union P4 { uint2 d; bf16_t h[4]; };
union P8 { uint4 d; bf16_t h[8]; };

__device__ __forceinline__ v8f wmma_bf16(const Frag& a, const Frag& b, v8f c) {
    return __builtin_amdgcn_wmma_f32_16x16x32_bf16(false, a.v, false, b.v,
                                                   (short)0, c, false, false);
}

// CDNA5 16-bit A/B fragment from a row-major bf16 tile row:
// elements 0..7 = k half*8..+7, elements 8..15 = k 16+half*8..+7 (half=lane>>4)
__device__ __forceinline__ Frag load_frag(const bf16_t* rowbase, int half) {
    Frag f;
    f.q[0] = *(const uint4*)(rowbase + 8 * half);
    f.q[1] = *(const uint4*)(rowbase + 16 + 8 * half);
    return f;
}
__device__ __forceinline__ Frag load_frag_lds(const unsigned short* rowbase, int half) {
    Frag f;
    f.q[0] = *(const uint4*)(rowbase + 8 * half);
    f.q[1] = *(const uint4*)(rowbase + 16 + 8 * half);
    return f;
}

// stage a 64x64 fp32 tile -> bf16 LDS (144B row stride), 128 threads
__device__ __forceinline__ void stage_tile(const float* __restrict__ src, int ld,
                                           unsigned short* __restrict__ dst,
                                           int r0, int kq) {
#pragma unroll
    for (int i = 0; i < 8; i++) {
        int row = r0 + i * 8;
        float4 a4 = *(const float4*)(src + (size_t)row * ld + kq * 4);
        P4 pa;
        pa.h[0] = (bf16_t)a4.x; pa.h[1] = (bf16_t)a4.y;
        pa.h[2] = (bf16_t)a4.z; pa.h[3] = (bf16_t)a4.w;
        *(uint2*)(dst + row * 72 + kq * 4) = pa.d;
    }
}

// ---------------------------------------------------------------------------
// GEMM: Y[M,N] = act((X[M,K] @ W[N,K]^T + bias) * scale) (+ residual)
// 128 threads = 4 waves, block tile 64x64, K-step 64, double-buffered LDS.
// OUT: 0 = f32 row-major, 1 = bf16 row-major, 2 = bf16 transposed [N][M].
// ---------------------------------------------------------------------------
template <int ACT, int OUT>
__global__ __launch_bounds__(128) void gemm_k(const float* __restrict__ X,
                                              const float* __restrict__ W,
                                              const float* __restrict__ bias,
                                              const float* __restrict__ resid,
                                              void* __restrict__ Yp,
                                              int M, int N, int K, float scale) {
    __shared__ __align__(16) unsigned short la[2][64 * 72];
    __shared__ __align__(16) unsigned short lw[2][64 * 72];

    const int tid  = threadIdx.x;
    const int lane = tid & 31;
    const int wid  = tid >> 5;
    const int half = lane >> 4;
    const int ln16 = lane & 15;

    const int bm = blockIdx.x * 64, bn = blockIdx.y * 64;
    const int wm = (wid >> 1) * 32, wn = (wid & 1) * 32;

    v8f c00 = {0,0,0,0,0,0,0,0}, c01 = c00, c10 = c00, c11 = c00;

    const int kq = tid & 15;   // float4 index along 64-wide k block
    const int r0 = tid >> 4;   // 0..7

    const float* Xb = X + (size_t)bm * K;
    const float* Wb = W + (size_t)bn * K;

    stage_tile(Xb, K, la[0], r0, kq);
    stage_tile(Wb, K, lw[0], r0, kq);
    __syncthreads();

    int cur = 0;
    for (int kb = 0; kb < K; kb += 64) {
        const int nxt = cur ^ 1;
        if (kb + 64 < K) {  // overlap next-tile staging with this tile's WMMAs
            stage_tile(Xb + kb + 64, K, la[nxt], r0, kq);
            stage_tile(Wb + kb + 64, K, lw[nxt], r0, kq);
        }
#pragma unroll
        for (int c = 0; c < 2; c++) {
            Frag a0 = load_frag_lds(la[cur] + (wm + ln16) * 72 + c * 32, half);
            Frag a1 = load_frag_lds(la[cur] + (wm + 16 + ln16) * 72 + c * 32, half);
            Frag b0 = load_frag_lds(lw[cur] + (wn + ln16) * 72 + c * 32, half);
            Frag b1 = load_frag_lds(lw[cur] + (wn + 16 + ln16) * 72 + c * 32, half);
            c00 = wmma_bf16(a0, b0, c00);
            c01 = wmma_bf16(a0, b1, c01);
            c10 = wmma_bf16(a1, b0, c10);
            c11 = wmma_bf16(a1, b1, c11);
        }
        __syncthreads();
        cur = nxt;
    }

    v8f* cc[2][2] = {{&c00, &c01}, {&c10, &c11}};
#pragma unroll
    for (int mi = 0; mi < 2; mi++)
#pragma unroll
        for (int ni = 0; ni < 2; ni++) {
            const int col = bn + wn + ni * 16 + ln16;
            const float bv = bias ? bias[col] : 0.f;
            if (OUT == 2) {
                // transposed bf16: lane owns 8 consecutive rows of one column
                P8 pk;
#pragma unroll
                for (int r = 0; r < 8; r++)
                    pk.h[r] = (bf16_t)(((*cc[mi][ni])[r] + bv) * scale);
                const int rowb = bm + wm + mi * 16 + half * 8;
                *(uint4*)((bf16_t*)Yp + (size_t)col * M + rowb) = pk.d;
            } else {
#pragma unroll
                for (int r = 0; r < 8; r++) {
                    const int row = bm + wm + mi * 16 + half * 8 + r;
                    float v = ((*cc[mi][ni])[r] + bv) * scale;
                    if (resid) v += resid[(size_t)row * N + col];
                    if (ACT == 1) v = fmaxf(v, 0.f);
                    if (OUT == 0)
                        ((float*)Yp)[(size_t)row * N + col] = v;
                    else
                        ((bf16_t*)Yp)[(size_t)row * N + col] = (bf16_t)v;
                }
            }
        }
}

// ---------------------------------------------------------------------------
// Flash attention, bf16 in / fp32 out. One wave per (head, 16-row Q block).
// All 16 global b128 loads issued at iteration top: V-fragment latency is
// hidden behind the softmax VALU work; K loads clause ahead of the S WMMAs.
// ---------------------------------------------------------------------------
__global__ __launch_bounds__(128) void flash_k(const bf16_t* __restrict__ Qh,
                                               const bf16_t* __restrict__ Kh,
                                               const bf16_t* __restrict__ Vt,
                                               float* __restrict__ O) {
    __shared__ __align__(16) unsigned short ldsp[4][16 * 40];

    const int tid  = threadIdx.x;
    const int lane = tid & 31;
    const int wid  = tid >> 5;
    const int gw   = blockIdx.x * 4 + wid;  // 0..1023
    const int head = gw >> 6;
    const int qb   = gw & 63;
    const int half = lane >> 4;
    const int ln16 = lane & 15;

    // Q B-fragments (lane = query column), loaded once
    const bf16_t* qrow = Qh + (size_t)(qb * 16 + ln16) * 1024 + head * 64;
    const Frag bq0 = load_frag(qrow, half);
    const Frag bq1 = load_frag(qrow + 32, half);

    Frag fones;
#pragma unroll
    for (int i = 0; i < 16; i++) fones.u[i] = 0x3F80;  // bf16 1.0

    float mrun = -1e30f;
    v8f o0 = {0,0,0,0,0,0,0,0}, o1 = o0, o2 = o0, o3 = o0, lacc = o0;
    v8f* op[4] = {&o0, &o1, &o2, &o3};

    unsigned short* pp = &ldsp[wid][0];
    const bf16_t* vbase = Vt + (size_t)(head * 64 + ln16) * 1024;
    const bf16_t* kbase_p = Kh + (size_t)ln16 * 1024 + head * 64;

    for (int jb = 0; jb < 32; jb++) {  // 32 keys per iteration
        const int kbase = jb * 32;

        // ---- issue all global loads up front (K for S, V for PV) ----
        Frag ak[2][2], bv[4];
#pragma unroll
        for (int t = 0; t < 2; t++) {
            const bf16_t* krow = kbase_p + (size_t)(kbase + t * 16) * 1024;
            ak[t][0] = load_frag(krow, half);
            ak[t][1] = load_frag(krow + 32, half);
        }
#pragma unroll
        for (int nt = 0; nt < 4; nt++)
            bv[nt] = load_frag(vbase + (size_t)nt * 16 * 1024 + kbase, half);

        // ---- S^T tiles (1/sqrt(dh) folded into Kh at projection) ----
        v8f st[2];
#pragma unroll
        for (int t = 0; t < 2; t++) {
            v8f acc = {0,0,0,0,0,0,0,0};
            acc = wmma_bf16(ak[t][0], bq0, acc);
            acc = wmma_bf16(ak[t][1], bq1, acc);
            st[t] = acc;
        }

        // per-lane max over this lane's 16 keys, then merge the two halves
        float m0 = fmaxf(fmaxf(fmaxf(st[0][0], st[0][1]), fmaxf(st[0][2], st[0][3])),
                         fmaxf(fmaxf(st[0][4], st[0][5]), fmaxf(st[0][6], st[0][7])));
        float m1 = fmaxf(fmaxf(fmaxf(st[1][0], st[1][1]), fmaxf(st[1][2], st[1][3])),
                         fmaxf(fmaxf(st[1][4], st[1][5]), fmaxf(st[1][6], st[1][7])));
        float mloc = fmaxf(m0, m1);
        mloc = fmaxf(mloc, __shfl_xor(mloc, 16, 32));
        const float mn = fmaxf(mrun, mloc);
        const float alpha = __expf(mrun - mn);
        mrun = mn;

        // P^T -> LDS as packed bf16 (one b128 store per tile): the store
        // transposes back to query-major rows for the PV A-fragment.
#pragma unroll
        for (int t = 0; t < 2; t++) {
            P8 pk;
#pragma unroll
            for (int r = 0; r < 8; r++)
                pk.h[r] = (bf16_t)__expf(st[t][r] - mn);
            *(uint4*)(pp + ln16 * 40 + t * 16 + half * 8) = pk.d;
        }

        // broadcast alpha from query-lane layout to C-layout rows
        float af[8];
#pragma unroll
        for (int r = 0; r < 8; r++) af[r] = __shfl(alpha, half * 8 + r, 32);
#pragma unroll
        for (int r = 0; r < 8; r++) {
            o0[r] *= af[r]; o1[r] *= af[r]; o2[r] *= af[r]; o3[r] *= af[r];
            lacc[r] *= af[r];
        }

        asm volatile("s_wait_dscnt 0" ::: "memory");  // P store -> A-frag load
        Frag ap = load_frag_lds(pp + ln16 * 40, half);

        // O += P @ V ; denominator += P @ ones
#pragma unroll
        for (int nt = 0; nt < 4; nt++)
            *op[nt] = wmma_bf16(ap, bv[nt], *op[nt]);
        lacc = wmma_bf16(ap, fones, lacc);
    }

#pragma unroll
    for (int r = 0; r < 8; r++) {
        const float inv = 1.f / lacc[r];
        const int row = qb * 16 + half * 8 + r;
#pragma unroll
        for (int nt = 0; nt < 4; nt++)
            O[(size_t)row * 1024 + head * 64 + nt * 16 + ln16] = (*op[nt])[r] * inv;
    }
}

// ---------------------------------------------------------------------------
// Row-wise LayerNorm over D=1024 (one block per row, 256 threads x float4)
// ---------------------------------------------------------------------------
__global__ __launch_bounds__(256) void layernorm_k(const float* __restrict__ X,
                                                   const float* __restrict__ g,
                                                   const float* __restrict__ b,
                                                   float* __restrict__ Y) {
    const int row = blockIdx.x, tid = threadIdx.x;
    __shared__ float red[8];

    float4 v = ((const float4*)(X + (size_t)row * 1024))[tid];
    float s = v.x + v.y + v.z + v.w;
#pragma unroll
    for (int m = 16; m >= 1; m >>= 1) s += __shfl_xor(s, m, 32);
    if ((tid & 31) == 0) red[tid >> 5] = s;
    __syncthreads();
    float tot = 0.f;
#pragma unroll
    for (int i = 0; i < 8; i++) tot += red[i];
    const float mean = tot * (1.f / 1024.f);
    __syncthreads();

    float4 d;
    d.x = v.x - mean; d.y = v.y - mean; d.z = v.z - mean; d.w = v.w - mean;
    float s2 = d.x * d.x + d.y * d.y + d.z * d.z + d.w * d.w;
#pragma unroll
    for (int m = 16; m >= 1; m >>= 1) s2 += __shfl_xor(s2, m, 32);
    if ((tid & 31) == 0) red[tid >> 5] = s2;
    __syncthreads();
    float tot2 = 0.f;
#pragma unroll
    for (int i = 0; i < 8; i++) tot2 += red[i];
    const float inv = rsqrtf(tot2 * (1.f / 1024.f) + 1e-5f);

    float4 gv = ((const float4*)g)[tid];
    float4 bv = ((const float4*)b)[tid];
    float4 o;
    o.x = d.x * inv * gv.x + bv.x;
    o.y = d.y * inv * gv.y + bv.y;
    o.z = d.z * inv * gv.z + bv.z;
    o.w = d.w * inv * gv.w + bv.w;
    ((float4*)(Y + (size_t)row * 1024))[tid] = o;
}

// Row L2-normalize (for cosine similarity)
__global__ __launch_bounds__(256) void rownorm_k(const float* __restrict__ X,
                                                 float* __restrict__ Y) {
    const int row = blockIdx.x, tid = threadIdx.x;
    __shared__ float red[8];
    float4 v = ((const float4*)(X + (size_t)row * 1024))[tid];
    float s2 = v.x * v.x + v.y * v.y + v.z * v.z + v.w * v.w;
#pragma unroll
    for (int m = 16; m >= 1; m >>= 1) s2 += __shfl_xor(s2, m, 32);
    if ((tid & 31) == 0) red[tid >> 5] = s2;
    __syncthreads();
    float tot = 0.f;
#pragma unroll
    for (int i = 0; i < 8; i++) tot += red[i];
    const float inv = rsqrtf(tot);
    float4 o;
    o.x = v.x * inv; o.y = v.y * inv; o.z = v.z * inv; o.w = v.w * inv;
    ((float4*)(Y + (size_t)row * 1024))[tid] = o;
}

// ---------------------------------------------------------------------------
// Host orchestration
// ---------------------------------------------------------------------------
extern "C" void kernel_launch(void* const* d_in, const int* in_sizes, int n_in,
                              void* d_out, int out_size, void* d_ws, size_t ws_size,
                              hipStream_t stream) {
    (void)in_sizes; (void)n_in; (void)out_size; (void)ws_size;
    auto F = [&](int i) -> const float* { return (const float*)d_in[i]; };

    struct AttnP { const float *Wq,*bq,*Wk,*bk,*Wv,*bv,*Wo,*bo; };
    struct FfnP  { const float *W1,*b1,*W2,*b2; };
    struct LnP   { const float *g,*b; };

    // pytree flatten order: dict keys sorted; 'W*' < 'b*' (ASCII).
    // top: shot_feature, params{dec, dec_norm, enc, enc_norm}
    // dec layer: ca(Wk,Wo,Wq,Wv,bk,bo,bq,bv), ff(W1,W2,b1,b2), n1(b,g), n2, n3, sa
    // enc layer: ff, n1, n2, sa
    const float* SRC = F(0);
    AttnP dsa[4], dca[4], esa[4];
    FfnP  dff[4], eff[4];
    LnP   dn1[4], dn2[4], dn3[4], en1[4], en2[4], encn, decn;
    int p = 1;
    auto rdA = [&](AttnP& A) {
        A.Wk = F(p + 0); A.Wo = F(p + 1); A.Wq = F(p + 2); A.Wv = F(p + 3);
        A.bk = F(p + 4); A.bo = F(p + 5); A.bq = F(p + 6); A.bv = F(p + 7);
        p += 8;
    };
    auto rdF = [&](FfnP& Fp) { Fp.W1 = F(p); Fp.W2 = F(p + 1); Fp.b1 = F(p + 2); Fp.b2 = F(p + 3); p += 4; };
    auto rdL = [&](LnP& L) { L.b = F(p); L.g = F(p + 1); p += 2; };
    for (int L = 0; L < 4; L++) { rdA(dca[L]); rdF(dff[L]); rdL(dn1[L]); rdL(dn2[L]); rdL(dn3[L]); rdA(dsa[L]); }
    rdL(decn);
    for (int L = 0; L < 4; L++) { rdF(eff[L]); rdL(en1[L]); rdL(en2[L]); rdA(esa[L]); }
    rdL(encn);

    // workspace carve-up: 8 MFloat fp32 + 3 x 1M bf16  (~38 MB total)
    float* W = (float*)d_ws;
    const size_t MB = 1024u * 1024u;
    float* Xb   = W + 0 * MB;
    float* Tb   = W + 1 * MB;
    float* Ob   = W + 2 * MB;
    float* Hb   = W + 3 * MB;   // 2M floats (1024x2048)
    float* MEMb = W + 5 * MB;
    float* Yb   = W + 6 * MB;
    float* XNb  = W + 7 * MB;
    bf16_t* Qh = (bf16_t*)(W + 8 * MB);
    bf16_t* Kh = Qh + MB;
    bf16_t* Vt = Kh + MB;

    auto gemm = [&](const float* X, const float* Wt, const float* bias,
                    const float* res, void* Yd, int M, int N, int K,
                    int act, int outm, float scale) {
        dim3 grid(M / 64, N / 64);
        if (outm == 1)
            gemm_k<0, 1><<<grid, 128, 0, stream>>>(X, Wt, bias, res, Yd, M, N, K, scale);
        else if (outm == 2)
            gemm_k<0, 2><<<grid, 128, 0, stream>>>(X, Wt, bias, res, Yd, M, N, K, scale);
        else if (act)
            gemm_k<1, 0><<<grid, 128, 0, stream>>>(X, Wt, bias, res, Yd, M, N, K, scale);
        else
            gemm_k<0, 0><<<grid, 128, 0, stream>>>(X, Wt, bias, res, Yd, M, N, K, scale);
    };
    auto ln = [&](const float* Xi, const LnP& L, float* Yd) {
        layernorm_k<<<1024, 256, 0, stream>>>(Xi, L.g, L.b, Yd);
    };
    auto mha = [&](const float* xq, const float* xkv, const AttnP& A, const LnP& L, float* xout) {
        gemm(xq,  A.Wq, A.bq, nullptr, Qh, 1024, 1024, 1024, 0, 1, 1.f);
        gemm(xkv, A.Wk, A.bk, nullptr, Kh, 1024, 1024, 1024, 0, 1, 0.125f);  // fold 1/sqrt(dh)
        gemm(xkv, A.Wv, A.bv, nullptr, Vt, 1024, 1024, 1024, 0, 2, 1.f);     // transposed
        flash_k<<<256, 128, 0, stream>>>(Qh, Kh, Vt, Ob);
        gemm(Ob, A.Wo, A.bo, xq, Tb, 1024, 1024, 1024, 0, 0, 1.f);  // + residual
        ln(Tb, L, xout);
    };
    auto ffn = [&](const float* x, const FfnP& Fp, const LnP& L, float* xout) {
        gemm(x,  Fp.W1, Fp.b1, nullptr, Hb, 1024, 2048, 1024, 1, 0, 1.f);  // ReLU
        gemm(Hb, Fp.W2, Fp.b2, x, Tb, 1024, 1024, 2048, 0, 0, 1.f);        // + residual
        ln(Tb, L, xout);
    };

    // encoder
    const float* x = SRC;
    for (int L = 0; L < 4; L++) {
        mha(x, x, esa[L], en1[L], Xb);
        ffn(Xb, eff[L], en2[L], Xb);
        x = Xb;
    }
    ln(Xb, encn, MEMb);

    // decoder (tgt == src)
    const float* y = SRC;
    for (int L = 0; L < 4; L++) {
        mha(y, y, dsa[L], dn1[L], Yb);          // self-attn
        mha(Yb, MEMb, dca[L], dn2[L], Yb);      // cross-attn
        ffn(Yb, dff[L], dn3[L], Yb);
        y = Yb;
    }

    float* out = (float*)d_out;
    ln(Yb, decn, out + MB);                         // attention_out -> d_out[1M..2M)
    rownorm_k<<<1024, 256, 0, stream>>>(out + MB, XNb);
    gemm(XNb, XNb, nullptr, nullptr, out, 1024, 1024, 1024, 0, 0, 1.f);  // xn @ xn^T
}
